// Time_Attention_687194767651
// MI455X (gfx1250) — matrix operand order repaired and checked by reading, not compiled, not run on hardware
//
#include <hip/hip_runtime.h>

// ---------------------------------------------------------------------------
// TimeSformer time-attention for MI455X (gfx1250, wave32, WMMA).
//
// Pipeline (all bf16 WMMA, f32 accumulate):
//   1) gemm_awt<fp32-in, bf16-out>:  qkv[50176,1536] = x[50176,512] @ w_qkv^T
//   2) attn_time: per (b,h,n) 16x16 softmax(QK^T/8) @ V      (6 WMMAs / wave)
//   3) gemm_awt<bf16-in, fp32-out>:  out = attn_out @ w_out^T + b_out
//      (A tile copy uses GLOBAL_LOAD_ASYNC_TO_LDS_B128 + s_wait_asynccnt)
//
// Roofline: ~107 GFLOP total; bf16 WMMA (~5 PFLOPS dense) -> ~21us compute,
// ~600MB HBM traffic -> ~26us @ 23.3TB/s: balanced, so bf16 (not the 8x
// slower K=4 fp32 WMMA) is the right precision. GEMMs are double-buffered in
// LDS so global->LDS staging of tile t+1 overlaps the 8 WMMAs of tile t.
// Workspace: qkv bf16 (154.1MB) + attn_out bf16 (51.4MB) = 205.6MB in d_ws.
// ---------------------------------------------------------------------------

typedef __bf16 bf16_t;
typedef __bf16 bf16x8  __attribute__((ext_vector_type(8)));
typedef __bf16 bf16x16 __attribute__((ext_vector_type(16)));
typedef float  v8f     __attribute__((ext_vector_type(8)));
typedef __attribute__((address_space(3))) bf16_t lds_bf16_t;

__device__ __forceinline__ v8f zerov8() {
    v8f z = {0.f, 0.f, 0.f, 0.f, 0.f, 0.f, 0.f, 0.f};
    return z;
}

__device__ __forceinline__ bf16x16 join8(bf16x8 lo, bf16x8 hi) {
    bf16x16 r;
#pragma unroll
    for (int i = 0; i < 8; ++i) { r[i] = lo[i]; r[i + 8] = hi[i]; }
    return r;
}

__device__ __forceinline__ bf16x8 cvt8(float4 a, float4 b) {
    bf16x8 r;
    r[0] = (bf16_t)a.x; r[1] = (bf16_t)a.y; r[2] = (bf16_t)a.z; r[3] = (bf16_t)a.w;
    r[4] = (bf16_t)b.x; r[5] = (bf16_t)b.y; r[6] = (bf16_t)b.z; r[7] = (bf16_t)b.w;
    return r;
}

__device__ __forceinline__ void load_f32_16(const float* g, float4& a0, float4& a1,
                                            float4& a2, float4& a3) {
    const float4* p = (const float4*)g;
    a0 = p[0]; a1 = p[1]; a2 = p[2]; a3 = p[3];
}

__device__ __forceinline__ void store_bf16_16(bf16_t* l, float4 a0, float4 a1,
                                              float4 a2, float4 a3) {
    *(bf16x8*)l       = cvt8(a0, a1);
    *(bf16x8*)(l + 8) = cvt8(a2, a3);
}

// CDNA5 async copy: 32 contiguous bytes global -> LDS, no VGPR data movement.
// Tracked by ASYNCcnt; pair with "s_wait_asynccnt 0" before the barrier.
__device__ __forceinline__ void async_copy32(bf16_t* lds_dst, const bf16_t* gsrc) {
    unsigned loff = (unsigned)(size_t)(lds_bf16_t*)lds_dst;   // LDS byte offset
    asm volatile(
        "global_load_async_to_lds_b128 %0, %1, off\n\t"
        "global_load_async_to_lds_b128 %0, %1, off offset:16"
        :: "v"(loff), "v"(gsrc)
        : "memory");
}

__device__ __forceinline__ void wait_async0() {
    asm volatile("s_wait_asynccnt 0x0" ::: "memory");
}

// C[M,N] = A[M,K] @ W[N,K]^T (+bias). 256 thr = 8 waves, block tile 128x128,
// wave grid 2(M)x4(N), per-wave 4x2 accum tiles of 16x16 -> 8 WMMA per k-step.
// LDS tiles bf16 [2][128][40] double-buffered (pad 8: 80B rows, 16B chunks).
template <bool A_IS_BF16, bool OUT_BF16>
__global__ __launch_bounds__(256) void gemm_awt(
    const float* __restrict__ Af, const bf16_t* __restrict__ Ab,
    const float* __restrict__ W, const float* __restrict__ bias,
    bf16_t* __restrict__ Cb, float* __restrict__ Cf,
    int M, int N, int K)
{
    __shared__ bf16_t Al[2][128 * 40];
    __shared__ bf16_t Wl[2][128 * 40];

    const int tid  = threadIdx.x;
    const int lane = tid & 31;
    const int wave = tid >> 5;
    const int wm   = wave >> 2;          // 0..1 : 64-row strip
    const int wn   = wave & 3;           // 0..3 : 32-col strip
    const int lm   = lane & 15;
    const int lh   = lane >> 4;
    const int m0   = blockIdx.y * 128;
    const int n0   = blockIdx.x * 128;

    // cooperative load mapping: 2 threads per row, 16 elements each
    const int r = tid >> 1;              // 0..127
    const int c = (tid & 1) << 4;        // 0 or 16

    v8f acc[4][2];
#pragma unroll
    for (int i = 0; i < 4; ++i)
#pragma unroll
        for (int j = 0; j < 2; ++j) acc[i][j] = zerov8();

    const int KT = K >> 5;

    // ---- prologue: stage k-tile 0 into buffer 0 ----
    if (A_IS_BF16) {
        async_copy32(&Al[0][r * 40 + c], Ab + (size_t)(m0 + r) * K + c);
    } else {
        float4 a0, a1, a2, a3;
        load_f32_16(Af + (size_t)(m0 + r) * K + c, a0, a1, a2, a3);
        store_bf16_16(&Al[0][r * 40 + c], a0, a1, a2, a3);
    }
    {
        float4 w0, w1, w2, w3;
        load_f32_16(W + (size_t)(n0 + r) * K + c, w0, w1, w2, w3);
        store_bf16_16(&Wl[0][r * 40 + c], w0, w1, w2, w3);
    }
    if (A_IS_BF16) wait_async0();
    __syncthreads();

    for (int kt = 0; kt < KT; ++kt) {
        const int  cur = kt & 1;
        const int  nxt = cur ^ 1;
        const bool hn  = (kt + 1) < KT;

        // ---- issue loads for k-tile kt+1 (overlap with WMMA below) ----
        float4 a0, a1, a2, a3, w0, w1, w2, w3;
        if (hn) {
            if (A_IS_BF16) {
                async_copy32(&Al[nxt][r * 40 + c],
                             Ab + (size_t)(m0 + r) * K + ((kt + 1) << 5) + c);
            } else {
                load_f32_16(Af + (size_t)(m0 + r) * K + ((kt + 1) << 5) + c,
                            a0, a1, a2, a3);
            }
            load_f32_16(W + (size_t)(n0 + r) * K + ((kt + 1) << 5) + c,
                        w0, w1, w2, w3);
        }
        if (kt + 2 < KT) {   // warm L2 two tiles ahead (global_prefetch_b8)
            __builtin_prefetch(W + (size_t)(n0 + r) * K + ((kt + 2) << 5) + c, 0, 1);
            if (A_IS_BF16)
                __builtin_prefetch(Ab + (size_t)(m0 + r) * K + ((kt + 2) << 5) + c, 0, 1);
            else
                __builtin_prefetch(Af + (size_t)(m0 + r) * K + ((kt + 2) << 5) + c, 0, 1);
        }

        // ---- fragments per ISA layouts (buffer cur) ----
        // A (16x32): lanes 0-15 hold K 0..7 / 16..23, lanes 16-31 K 8..15 / 24..31
        bf16x16 afr[4];
#pragma unroll
        for (int i = 0; i < 4; ++i) {
            const int row = wm * 64 + i * 16 + lm;
            bf16x8 lo = *(const bf16x8*)&Al[cur][row * 40 + lh * 8];
            bf16x8 hi = *(const bf16x8*)&Al[cur][row * 40 + 16 + lh * 8];
            afr[i] = join8(lo, hi);
        }
        // B (32x16): lane = column; lanes 0-15 K 0..7 / 16..23, lanes 16-31 K 8..15 / 24..31
        bf16x16 bfr[2];
#pragma unroll
        for (int j = 0; j < 2; ++j) {
            const int nrow = wn * 32 + j * 16 + lm;
            bf16x8 lo = *(const bf16x8*)&Wl[cur][nrow * 40 + lh * 8];
            bf16x8 hi = *(const bf16x8*)&Wl[cur][nrow * 40 + 16 + lh * 8];
            bfr[j] = join8(lo, hi);
        }
#pragma unroll
        for (int i = 0; i < 4; ++i)
#pragma unroll
            for (int j = 0; j < 2; ++j)
                acc[i][j] = __builtin_amdgcn_wmma_f32_16x16x32_bf16(
                    false, afr[i], false, bfr[j], (short)0, acc[i][j], false, false);

        // ---- drain staging regs into buffer nxt ----
        if (hn) {
            if (!A_IS_BF16) store_bf16_16(&Al[nxt][r * 40 + c], a0, a1, a2, a3);
            store_bf16_16(&Wl[nxt][r * 40 + c], w0, w1, w2, w3);
        }
        if (A_IS_BF16) wait_async0();
        __syncthreads();
    }

    // ---- epilogue: C layout = VGPR v holds row (v + 8*lh), col = lm ----
#pragma unroll
    for (int i = 0; i < 4; ++i) {
        const int grow0 = m0 + wm * 64 + i * 16 + lh * 8;
#pragma unroll
        for (int j = 0; j < 2; ++j) {
            const int gcol = n0 + wn * 32 + j * 16 + lm;
#pragma unroll
            for (int vv = 0; vv < 8; ++vv) {
                const size_t idx = (size_t)(grow0 + vv) * N + gcol;
                if (OUT_BF16) Cb[idx] = (bf16_t)acc[i][j][vv];
                else          Cf[idx] = acc[i][j][vv] + bias[gcol];
            }
        }
    }
}

// One wave per (b,h,n): sim = Q K^T over DH=64 (2 WMMAs), softmax over F via
// 16-lane shuffle reductions, attn relayout via LDS, out = attn V (4 WMMAs,
// K half zero-padded since F=16 < 32).
__global__ __launch_bounds__(256) void attn_time(
    const bf16_t* __restrict__ qkv, bf16_t* __restrict__ aout)
{
    __shared__ bf16_t pl[8][16 * 24];   // per-wave 16x16 attn tile (pad 24)

    const int tid = threadIdx.x, lane = tid & 31, wave = tid >> 5;
    const int tile = blockIdx.x * 8 + wave;       // 3136*8 = 25088 tiles exact
    const int b   = tile / (8 * 196);
    const int rem = tile - b * (8 * 196);
    const int h   = rem / 196;
    const int n   = rem - h * 196;
    const int lm  = lane & 15, lh = lane >> 4;

    const size_t frow = (size_t)196 * 1536;       // qkv row stride per frame
    const bf16_t* qb = qkv + ((size_t)b * 3136 + n) * 1536 + h * 64;
    const bf16_t* kb = qb + 512;
    const bf16_t* vb = qb + 1024;

    // ---- sim = Q K^T ----
    v8f sim = zerov8();
#pragma unroll
    for (int ks = 0; ks < 2; ++ks) {
        const bf16_t* qp = qb + (size_t)lm * frow + ks * 32;
        bf16x16 af = join8(*(const bf16x8*)(qp + lh * 8),
                           *(const bf16x8*)(qp + 16 + lh * 8));
        const bf16_t* kp = kb + (size_t)lm * frow + ks * 32;
        bf16x16 bf = join8(*(const bf16x8*)(kp + lh * 8), *(const bf16x8*)(kp + 16 + lh * 8));
        sim = __builtin_amdgcn_wmma_f32_16x16x32_bf16(
            false, af, false, bf, (short)0, sim, false, false);
    }

    // ---- softmax over g (columns <-> lanes within 16-lane half) ----
#pragma unroll
    for (int i = 0; i < 8; ++i) {
        float s = sim[i] * 0.125f;                // DH^-0.5
        float mx = s;
#pragma unroll
        for (int d = 1; d < 16; d <<= 1) mx = fmaxf(mx, __shfl_xor(mx, d, 32));
        float e = __expf(s - mx);
        float sum = e;
#pragma unroll
        for (int d = 1; d < 16; d <<= 1) sum += __shfl_xor(sum, d, 32);
        pl[wave][(i + lh * 8) * 24 + lm] = (bf16_t)(e / sum);
    }
    __syncthreads();

    // ---- relayout attn into A fragment (K=16 valid, 16..31 zero) ----
    bf16x8 z8;
#pragma unroll
    for (int i = 0; i < 8; ++i) z8[i] = (bf16_t)0.f;
    bf16x16 af2 = join8(*(const bf16x8*)&pl[wave][lm * 24 + lh * 8], z8);

    // ---- out = attn @ V ----
    v8f o[4];
#pragma unroll
    for (int j = 0; j < 4; ++j) o[j] = zerov8();
#pragma unroll
    for (int j = 0; j < 4; ++j) {
        bf16x16 vf;
        const bf16_t* vp = vb + j * 16 + lm;      // column d = j*16 + lane
#pragma unroll
        for (int kk = 0; kk < 8; ++kk) vf[kk] = vp[(size_t)(kk + lh * 8) * frow];
        {                                          // K 16..31 = 0
#pragma unroll
            for (int kk = 8; kk < 16; ++kk) vf[kk] = (bf16_t)0.f;
        }
        o[j] = __builtin_amdgcn_wmma_f32_16x16x32_bf16(
            false, af2, false, vf, (short)0, o[j], false, false);
    }

    // ---- store: row f = vv + 8*lh, col = h*64 + j*16 + lm ----
    const size_t orow = (size_t)196 * 512;
    const size_t ob   = ((size_t)b * 3136 + n) * 512 + h * 64;
#pragma unroll
    for (int j = 0; j < 4; ++j)
#pragma unroll
        for (int vv = 0; vv < 8; ++vv)
            aout[ob + (size_t)(vv + lh * 8) * orow + j * 16 + lm] =
                (bf16_t)o[j][vv];
}

extern "C" void kernel_launch(void* const* d_in, const int* in_sizes, int n_in,
                              void* d_out, int out_size, void* d_ws, size_t ws_size,
                              hipStream_t stream) {
    (void)in_sizes; (void)n_in; (void)out_size; (void)ws_size;
    const float* x     = (const float*)d_in[0];   // [16, 3136, 512]
    const float* w_qkv = (const float*)d_in[1];   // [1536, 512]
    const float* w_out = (const float*)d_in[2];   // [512, 512]
    const float* b_out = (const float*)d_in[3];   // [512]
    float* out = (float*)d_out;                   // [16, 3136, 512]

    bf16_t* qkv  = (bf16_t*)d_ws;                        // [50176, 1536] bf16
    bf16_t* aout = qkv + (size_t)50176 * 1536;           // [50176, 512]  bf16

    const dim3 blk(256);
    // 1) qkv = x @ w_qkv^T  (M=50176, N=1536, K=512)
    gemm_awt<false, true><<<dim3(12, 392), blk, 0, stream>>>(
        x, nullptr, w_qkv, nullptr, qkv, nullptr, 50176, 1536, 512);
    // 2) time attention over F=16 per (b,h,n)
    attn_time<<<dim3(3136), blk, 0, stream>>>(qkv, aout);
    // 3) out = attn_out @ w_out^T + b_out  (M=50176, N=512, K=512)
    gemm_awt<true, false><<<dim3(4, 392), blk, 0, stream>>>(
        nullptr, aout, w_out, b_out, nullptr, out, 50176, 512, 512);
}
